// KCenter_28656021799325
// MI455X (gfx1250) — matrix operand order repaired and checked by reading, not compiled, not run on hardware
//
#include <hip/hip_runtime.h>

// ---------------------------------------------------------------------------
// k-center farthest-point step on MI455X (gfx1250, wave32, WMMA).
//
// D[i,j] = sqrt(max(na_i - 2 A_i.B_j + nb_j, 0));  out = (argmax_i min_j D, max)
// GEMM part (256 GFLOP) runs as bf16 split (hi+lo limbs, 3 WMMAs per k-chunk)
// on v_wmma_f32_16x16x32_bf16 with f32 accumulation (~1e-5 rel. error).
// ---------------------------------------------------------------------------

typedef __attribute__((ext_vector_type(16))) __bf16 v16bf;
typedef __attribute__((ext_vector_type(8)))  __bf16 v8bf;
typedef __attribute__((ext_vector_type(4)))  __bf16 v4bf;
typedef __attribute__((ext_vector_type(8)))  float  v8f;

#define KDIM     512
#define BR       128          // A rows per block (8 waves x 16 rows)
#define JCH      128          // B rows (distance columns) per chunk
#define KC       32           // K depth staged in LDS per step (= one WMMA k)
#define LDS_LD   (KC + 8)     // padded bf16 row stride: 40 elems = 20 dwords
#define NTHREADS 256

static __device__ __forceinline__ float warp_min16(float v) {
  // min across each 16-lane half of the wave32 (halves stay independent)
  v = fminf(v, __shfl_xor(v, 1, 32));
  v = fminf(v, __shfl_xor(v, 2, 32));
  v = fminf(v, __shfl_xor(v, 4, 32));
  v = fminf(v, __shfl_xor(v, 8, 32));
  return v;
}

// one wave per row: sum of squares of 512 f32
__global__ void row_norms(const float* __restrict__ X, float* __restrict__ out,
                          int nrows) {
  const int lane = threadIdx.x & 31;
  const int row  = (blockIdx.x * blockDim.x + threadIdx.x) >> 5;
  if (row >= nrows) return;
  const float4* r4 = (const float4*)(X + (size_t)row * KDIM);
  float s = 0.f;
#pragma unroll
  for (int i = 0; i < KDIM / 128; ++i) {      // 128 float4 / 32 lanes
    float4 v = r4[lane + 32 * i];
    s += v.x * v.x + v.y * v.y + v.z * v.z + v.w * v.w;
  }
#pragma unroll
  for (int off = 16; off; off >>= 1) s += __shfl_xor(s, off, 32);
  if (lane == 0) out[row] = s;
}

// split f32 -> bf16 hi + bf16 lo limbs and store 4 contiguous elems to LDS
static __device__ __forceinline__ void cvt_store4(__bf16* hi, __bf16* lo,
                                                  int idx, float4 v) {
  float xs[4] = {v.x, v.y, v.z, v.w};
  v4bf h, l;
#pragma unroll
  for (int i = 0; i < 4; ++i) {
    __bf16 hb = (__bf16)xs[i];
    h[i] = hb;
    l[i] = (__bf16)(xs[i] - (float)hb);
  }
  *(v4bf*)(hi + idx) = h;
  *(v4bf*)(lo + idx) = l;
}

__global__ __launch_bounds__(NTHREADS) void kcenter_main(
    const float* __restrict__ A, const float* __restrict__ B,
    const float* __restrict__ na, const float* __restrict__ nb,
    float* __restrict__ blkVal, int* __restrict__ blkIdx, int N, int M) {
  __shared__ __bf16 Ah[BR * LDS_LD];
  __shared__ __bf16 Al[BR * LDS_LD];
  __shared__ __bf16 Bh[JCH * LDS_LD];
  __shared__ __bf16 Bl[JCH * LDS_LD];
  __shared__ float  redV[BR];
  __shared__ int    redI[BR];

  const int tid   = threadIdx.x;
  const int lane  = tid & 31;
  const int wave  = tid >> 5;
  const int mrow  = lane & 15;   // M (A frag) / N (B frag) index within tile
  const int lhalf = lane >> 4;   // lane half selects K sub-range per ISA layout
  const int i0    = blockIdx.x * BR;
  const int arow  = wave * 16 + mrow;  // local A row feeding this lane's frag

  float vmin[8];                 // running min_j of (nb_j - 2 s_ij), acc layout
#pragma unroll
  for (int r = 0; r < 8; ++r) vmin[r] = __builtin_inff();

  const int njc = (M + JCH - 1) / JCH;
  for (int jc = 0; jc < njc; ++jc) {
    const int j0 = jc * JCH;
    v8f acc[8] = {};             // 8 column sub-tiles of 16, rows = wave's 16

    for (int kk = 0; kk < KDIM; kk += KC) {
      __syncthreads();
      // ---- stage A[BR x KC] as bf16 hi/lo -------------------------------
#pragma unroll
      for (int p = 0; p < (BR * KC / 4) / NTHREADS; ++p) {
        int f4 = p * NTHREADS + tid;
        int r  = f4 >> 3;                 // KC/4 = 8 float4 per row
        int c  = (f4 & 7) << 2;
        int gr = i0 + r; gr = gr < N ? gr : N - 1;   // clamp row tail
        float4 v = *(const float4*)(A + (size_t)gr * KDIM + kk + c);
        cvt_store4(Ah, Al, r * LDS_LD + c, v);
      }
      // ---- stage B[JCH x KC] as bf16 hi/lo ------------------------------
#pragma unroll
      for (int p = 0; p < (JCH * KC / 4) / NTHREADS; ++p) {
        int f4 = p * NTHREADS + tid;
        int r  = f4 >> 3;
        int c  = (f4 & 7) << 2;
        int gj = j0 + r; gj = gj < M ? gj : M - 1;   // clamp col tail
        float4 v = *(const float4*)(B + (size_t)gj * KDIM + kk + c);
        cvt_store4(Bh, Bl, r * LDS_LD + c, v);
      }
      __syncthreads();

      // ---- one 16x16x32 bf16 k-chunk ------------------------------------
      // A(16x32) frag: lanes<16 hold K 0..7 & 16..23, lanes>=16 K 8..15 & 24..31
      v8bf a0h = *(const v8bf*)&Ah[arow * LDS_LD + 8 * lhalf];
      v8bf a1h = *(const v8bf*)&Ah[arow * LDS_LD + 16 + 8 * lhalf];
      v8bf a0l = *(const v8bf*)&Al[arow * LDS_LD + 8 * lhalf];
      v8bf a1l = *(const v8bf*)&Al[arow * LDS_LD + 16 + 8 * lhalf];
      v16bf ahi = __builtin_shufflevector(a0h, a1h, 0,1,2,3,4,5,6,7,8,9,10,11,12,13,14,15);
      v16bf alo = __builtin_shufflevector(a0l, a1l, 0,1,2,3,4,5,6,7,8,9,10,11,12,13,14,15);

#pragma unroll
      for (int t = 0; t < 8; ++t) {
        // B(32x16) frag: lane holds N=mrow, K 8*lhalf..+7 & 16+8*lhalf..+7
        const int brow = t * 16 + mrow;
        v8bf b0h = *(const v8bf*)&Bh[brow * LDS_LD + 8 * lhalf];
        v8bf b1h = *(const v8bf*)&Bh[brow * LDS_LD + 16 + 8 * lhalf];
        v8bf b0l = *(const v8bf*)&Bl[brow * LDS_LD + 8 * lhalf];
        v8bf b1l = *(const v8bf*)&Bl[brow * LDS_LD + 16 + 8 * lhalf];
        v16bf bhi = __builtin_shufflevector(b0h, b1h, 0,1,2,3,4,5,6,7,8,9,10,11,12,13,14,15);
        v16bf blo = __builtin_shufflevector(b0l, b1l, 0,1,2,3,4,5,6,7,8,9,10,11,12,13,14,15);
        // s += ahi*bhi + ahi*blo + alo*bhi   (drop lo*lo term)
        acc[t] = __builtin_amdgcn_wmma_f32_16x16x32_bf16(false, ahi, false, bhi,
                                                         (short)0, acc[t], false, false);
        acc[t] = __builtin_amdgcn_wmma_f32_16x16x32_bf16(false, ahi, false, blo,
                                                         (short)0, acc[t], false, false);
        acc[t] = __builtin_amdgcn_wmma_f32_16x16x32_bf16(false, alo, false, bhi,
                                                         (short)0, acc[t], false, false);
      }
    }

    // ---- epilogue: fold this column chunk into running min --------------
#pragma unroll
    for (int t = 0; t < 8; ++t) {
      int   col = j0 + t * 16 + mrow;
      float nbv = (col < M) ? nb[col] : __builtin_inff();   // mask col tail
#pragma unroll
      for (int r = 0; r < 8; ++r)
        vmin[r] = fminf(vmin[r], nbv - 2.0f * acc[t][r]);
    }
  }

  // per-row min across the 16 lanes of each half; C/D layout: VGPR r holds
  // row r (lanes 0-15) and row r+8 (lanes 16-31)
#pragma unroll
  for (int r = 0; r < 8; ++r) {
    float v = warp_min16(vmin[r]);
    if (mrow == 0) {
      int lrow = wave * 16 + lhalf * 8 + r;
      int grow = i0 + lrow;
      float d;
      if (grow < N) {
        d = sqrtf(fmaxf(na[grow] + v, 0.0f));
      } else {
        d = -__builtin_inff();      // masked rows never win the max
      }
      redV[lrow] = d;
      redI[lrow] = grow;
    }
  }
  __syncthreads();

  // block argmax (first-index tie-break, matches jnp.argmax)
  if (wave == 0) {
    float bv = -__builtin_inff();
    int   bi = 0x7fffffff;
#pragma unroll
    for (int p = 0; p < BR / 32; ++p) {
      float v = redV[p * 32 + lane];
      int   i = redI[p * 32 + lane];
      if (v > bv || (v == bv && i < bi)) { bv = v; bi = i; }
    }
#pragma unroll
    for (int off = 16; off; off >>= 1) {
      float ov = __shfl_xor(bv, off, 32);
      int   oi = __shfl_xor(bi, off, 32);
      if (ov > bv || (ov == bv && oi < bi)) { bv = ov; bi = oi; }
    }
    if (lane == 0) { blkVal[blockIdx.x] = bv; blkIdx[blockIdx.x] = bi; }
  }
}

__global__ void kcenter_final(const float* __restrict__ blkVal,
                              const int* __restrict__ blkIdx, int nblk,
                              float* __restrict__ out) {
  const int lane = threadIdx.x & 31;
  float bv = -__builtin_inff();
  int   bi = 0x7fffffff;
  for (int p = lane; p < nblk; p += 32) {
    float v = blkVal[p];
    int   i = blkIdx[p];
    if (v > bv || (v == bv && i < bi)) { bv = v; bi = i; }
  }
#pragma unroll
  for (int off = 16; off; off >>= 1) {
    float ov = __shfl_xor(bv, off, 32);
    int   oi = __shfl_xor(bi, off, 32);
    if (ov > bv || (ov == bv && oi < bi)) { bv = ov; bi = oi; }
  }
  if (lane == 0) {
    out[0] = (float)bi;   // argmax index
    out[1] = bv;          // max of per-row min distance
  }
}

extern "C" void kernel_launch(void* const* d_in, const int* in_sizes, int n_in,
                              void* d_out, int out_size, void* d_ws, size_t ws_size,
                              hipStream_t stream) {
  const float* A = (const float*)d_in[0];
  const float* B = (const float*)d_in[1];
  const int N = in_sizes[0] / KDIM;    // 50000
  const int M = in_sizes[1] / KDIM;    // 5000

  // workspace carve: na[N] | nb[M] | blkVal[nblk] | blkIdx[nblk]
  float* na   = (float*)d_ws;
  float* nb   = na + N;
  const int nblk = (N + BR - 1) / BR;
  float* bVal = nb + M;
  int*   bIdx = (int*)(bVal + nblk);

  row_norms<<<(N + 7) / 8, NTHREADS, 0, stream>>>(A, na, N);
  row_norms<<<(M + 7) / 8, NTHREADS, 0, stream>>>(B, nb, M);
  kcenter_main<<<nblk, NTHREADS, 0, stream>>>(A, B, na, nb, bVal, bIdx, N, M);
  kcenter_final<<<1, 32, 0, stream>>>(bVal, bIdx, nblk, (float*)d_out);
}
